// GAT_23390391894786
// MI455X (gfx1250) — matrix operand order repaired and checked by reading, not compiled, not run on hardware
//
#include <hip/hip_runtime.h>
#include <hip/hip_bf16.h>

typedef __attribute__((ext_vector_type(16))) __bf16          v16bf;
typedef __attribute__((ext_vector_type(16))) unsigned short  v16us;
typedef __attribute__((ext_vector_type(8)))  float           v8f;
typedef __attribute__((ext_vector_type(8)))  unsigned int    v8u;
typedef __attribute__((ext_vector_type(2)))  float           f32x2;

// ---------- helpers ----------
__device__ __forceinline__ unsigned short f2bf_rne(float f) {
    unsigned u = __float_as_uint(f);
    unsigned r = u + 0x7FFFu + ((u >> 16) & 1u);
    return (unsigned short)(r >> 16);
}
__device__ __forceinline__ float bf2f(unsigned short h) {
    return __uint_as_float(((unsigned)h) << 16);
}
// order-preserving f32 <-> u32 (for atomicMax on floats)
__device__ __forceinline__ unsigned enc_f32(float f) {
    unsigned u = __float_as_uint(f);
    return (u & 0x80000000u) ? ~u : (u | 0x80000000u);
}
__device__ __forceinline__ float dec_f32(unsigned u) {
    return __uint_as_float((u & 0x80000000u) ? (u & 0x7FFFFFFFu) : ~u);
}

// ---------- generic u32 fill ----------
__global__ void fill_u32(unsigned* __restrict__ p, unsigned val, size_t n) {
    size_t i = (size_t)blockIdx.x * blockDim.x + threadIdx.x;
    if (i < n) p[i] = val;
}

// ---------- pack W[K,NN] (row-major f32) into per-lane bf16 hi/lo WMMA B layout ----------
// entry layout: [(kt*NT + nt)*32 + lane][16 halves]; B 32x16: n = lane&15,
// k = kt*32 + ((e<8) ? (8*(lane>>4) + e) : (16 + 8*(lane>>4) + e - 8))      (per CDNA5 ISA B-matrix layout)
__global__ void pack_w(const float* __restrict__ W, unsigned short* __restrict__ hi,
                       unsigned short* __restrict__ lo, int K, int NN) {
    int idx = blockIdx.x * blockDim.x + threadIdx.x;
    int total = (K >> 5) * (NN >> 4) * 32 * 16;
    if (idx >= total) return;
    int e    = idx & 15;
    int lane = (idx >> 4) & 31;
    int tile = idx >> 9;
    int NT = NN >> 4;
    int kt = tile / NT, nt = tile % NT;
    int n = nt * 16 + (lane & 15);
    int k = kt * 32 + ((e < 8) ? (((lane < 16) ? 0 : 8) + e) : (16 + ((lane < 16) ? 0 : 8) + (e - 8)));
    float f = W[(size_t)k * NN + n];
    unsigned short h = f2bf_rne(f);
    hi[idx] = h;
    lo[idx] = f2bf_rne(f - bf2f(h));
}

// ---------- GEMM: C[M,NN] = A[M,K] * W, bf16x3 split via v_wmma_f32_16x16x32_bf16 ----------
// one wave -> 16(M) x 64(N) output strip, 4 accumulators
__global__ void gemm_bf16x3(const float* __restrict__ A,
                            const unsigned short* __restrict__ Whi,
                            const unsigned short* __restrict__ Wlo,
                            float* __restrict__ C,
                            int K, int NN, int nunits) {
    int lane = threadIdx.x & 31;
    int wave = threadIdx.x >> 5;
    int unit = blockIdx.x * (blockDim.x >> 5) + wave;
    if (unit >= nunits) return;
    int nstrips = NN >> 6;
    int mt = unit / nstrips;
    int ns = unit % nstrips;
    int NT = NN >> 4;

    int m0 = mt * 16 + (lane & 15);
    int kb = (lane < 16) ? 0 : 8;          // A-matrix K interleave base (ISA table)
    const float* arow = A + (size_t)m0 * K;

    v8f acc0 = {}, acc1 = {}, acc2 = {}, acc3 = {};

    int kt_end = K >> 5;
    for (int kt = 0; kt < kt_end; ++kt) {
        __builtin_prefetch(arow + kt * 32 + 128, 0, 1);  // global_prefetch_b8
        // --- load A tile (16x32 per ISA A layout), split fp32 -> bf16 hi/lo ---
        v16us ahu, alu;
#pragma unroll
        for (int v = 0; v < 8; ++v) {
            int k = kt * 32 + ((v < 4) ? (kb + 2 * v) : (16 + kb + 2 * (v - 4)));
            f32x2 f = *(const f32x2*)(arow + k);
            unsigned short h0 = f2bf_rne(f.x), h1 = f2bf_rne(f.y);
            ahu[2 * v]     = h0;
            ahu[2 * v + 1] = h1;
            alu[2 * v]     = f2bf_rne(f.x - bf2f(h0));
            alu[2 * v + 1] = f2bf_rne(f.y - bf2f(h1));
        }
        v16bf ahi = __builtin_bit_cast(v16bf, ahu);
        v16bf alo = __builtin_bit_cast(v16bf, alu);

#pragma unroll
        for (int t = 0; t < 4; ++t) {
            size_t base = ((size_t)(kt * NT + (ns * 4 + t)) * 32 + lane) * 16;
            v16bf bhi = __builtin_bit_cast(v16bf, *(const v8u*)(Whi + base));
            v16bf blo = __builtin_bit_cast(v16bf, *(const v8u*)(Wlo + base));
            v8f a = (t == 0) ? acc0 : (t == 1) ? acc1 : (t == 2) ? acc2 : acc3;
            a = __builtin_amdgcn_wmma_f32_16x16x32_bf16(false, ahi, false, bhi, (short)0, a, false, false);
            a = __builtin_amdgcn_wmma_f32_16x16x32_bf16(false, ahi, false, blo, (short)0, a, false, false);
            a = __builtin_amdgcn_wmma_f32_16x16x32_bf16(false, alo, false, bhi, (short)0, a, false, false);
            if (t == 0) acc0 = a; else if (t == 1) acc1 = a; else if (t == 2) acc2 = a; else acc3 = a;
        }
    }

    // --- epilogue: C/D layout m = v + 8*(lane>=16), n = lane&15 ---
    int mrow = mt * 16 + ((lane < 16) ? 0 : 8);
    int nc0  = ns * 64 + (lane & 15);
#pragma unroll
    for (int t = 0; t < 4; ++t) {
        v8f a = (t == 0) ? acc0 : (t == 1) ? acc1 : (t == 2) ? acc2 : acc3;
#pragma unroll
        for (int v = 0; v < 8; ++v)
            C[(size_t)(mrow + v) * NN + nc0 + t * 16] = a[v];
    }
}

// ---------- attention dot products: a_src[n,h] = sum_c h[n,h,c]*att_src[h,c] ----------
__global__ void attn_dots(const float* __restrict__ hmat, const float* __restrict__ att_s,
                          const float* __restrict__ att_d, float* __restrict__ asrc,
                          float* __restrict__ adst, int N, int heads, int Cc) {
    int lane = threadIdx.x & 31;
    int wave = threadIdx.x >> 5;
    int n = blockIdx.x * (blockDim.x >> 5) + wave;
    if (n >= N) return;
    int HC = heads * Cc;
    for (int hh = 0; hh < heads; ++hh) {
        float s = 0.f, d = 0.f;
        for (int c = lane; c < Cc; c += 32) {
            float hv = hmat[(size_t)n * HC + hh * Cc + c];
            s += hv * att_s[hh * Cc + c];
            d += hv * att_d[hh * Cc + c];
        }
#pragma unroll
        for (int off = 16; off > 0; off >>= 1) {
            s += __shfl_down(s, off, 32);
            d += __shfl_down(d, off, 32);
        }
        if (lane == 0) { asrc[n * heads + hh] = s; adst[n * heads + hh] = d; }
    }
}

// ---------- edge pass 1: segment max (sortable-u32 atomicMax) ----------
__global__ void edge_max(const int* __restrict__ ei, const float* __restrict__ asrc,
                         const float* __restrict__ adst, unsigned* __restrict__ emax,
                         int E, int N, int heads, float slope) {
    int e = blockIdx.x * blockDim.x + threadIdx.x;
    if (e >= E + N) return;
    int s = (e < E) ? ei[e]     : (e - E);
    int d = (e < E) ? ei[E + e] : (e - E);
    for (int hh = 0; hh < heads; ++hh) {
        float v = asrc[s * heads + hh] + adst[d * heads + hh];
        v = (v > 0.f) ? v : v * slope;
        atomicMax(&emax[d * heads + hh], enc_f32(v));
    }
}

// ---------- edge pass 2: denom = segment_sum(exp(e - max)) ----------
__global__ void edge_denom(const int* __restrict__ ei, const float* __restrict__ asrc,
                           const float* __restrict__ adst, const unsigned* __restrict__ emax,
                           float* __restrict__ denom, int E, int N, int heads, float slope) {
    int e = blockIdx.x * blockDim.x + threadIdx.x;
    if (e >= E + N) return;
    int s = (e < E) ? ei[e]     : (e - E);
    int d = (e < E) ? ei[E + e] : (e - E);
    for (int hh = 0; hh < heads; ++hh) {
        float v = asrc[s * heads + hh] + adst[d * heads + hh];
        v = (v > 0.f) ? v : v * slope;
        atomicAdd(&denom[d * heads + hh], __expf(v - dec_f32(emax[d * heads + hh])));
    }
}

// ---------- edge pass 3: out[dst] += alpha * h[src]  (thread per edge-channel) ----------
__global__ void edge_msg(const int* __restrict__ ei, const float* __restrict__ asrc,
                         const float* __restrict__ adst, const unsigned* __restrict__ emax,
                         const float* __restrict__ denom, const float* __restrict__ hsrc,
                         float* __restrict__ out, int E, int N, int heads, int Cc, float slope) {
    size_t idx = (size_t)blockIdx.x * blockDim.x + threadIdx.x;
    int c = (int)(idx % Cc);
    size_t e = idx / Cc;
    if (e >= (size_t)(E + N)) return;
    int s = (e < (size_t)E) ? ei[e]     : (int)(e - E);
    int d = (e < (size_t)E) ? ei[E + e] : (int)(e - E);
    int HC = heads * Cc;
    for (int hh = 0; hh < heads; ++hh) {
        float v = asrc[s * heads + hh] + adst[d * heads + hh];
        v = (v > 0.f) ? v : v * slope;
        float alpha = __expf(v - dec_f32(emax[d * heads + hh])) / denom[d * heads + hh];
        atomicAdd(&out[(size_t)d * HC + hh * Cc + c],
                  hsrc[(size_t)s * HC + hh * Cc + c] * alpha);
    }
}

// ---------- bias (+optional ELU), in place ----------
__global__ void bias_act(float* __restrict__ io, const float* __restrict__ bias,
                         size_t total, int HC, int do_elu) {
    size_t i = (size_t)blockIdx.x * blockDim.x + threadIdx.x;
    if (i >= total) return;
    float v = io[i] + bias[i % HC];
    if (do_elu) v = (v > 0.f) ? v : (__expf(v) - 1.f);
    io[i] = v;
}

// =========================== launcher ===========================
extern "C" void kernel_launch(void* const* d_in, const int* in_sizes, int n_in,
                              void* d_out, int out_size, void* d_ws, size_t ws_size,
                              hipStream_t stream) {
    const float* x        = (const float*)d_in[0];
    const int*   ei       = (const int*)  d_in[1];
    const float* W1       = (const float*)d_in[2];
    const float* att_s1   = (const float*)d_in[3];
    const float* att_d1   = (const float*)d_in[4];
    const float* bias1    = (const float*)d_in[5];
    const float* W2       = (const float*)d_in[6];
    const float* att_s2   = (const float*)d_in[7];
    const float* att_d2   = (const float*)d_in[8];
    const float* bias2    = (const float*)d_in[9];

    const int HC1   = in_sizes[5];            // 256 = heads*hidden
    const int m     = in_sizes[2] / HC1;      // 512
    const int C2    = in_sizes[9];            // 64
    const int N     = in_sizes[0] / m;        // 50000
    const int E     = in_sizes[1] / 2;        // 800000
    const int H1    = 4, Ch1 = HC1 / H1;      // 4 heads x 64
    const int H2    = 1;
    const float slope = 0.2f;

    // ---- workspace carve-out ----
    char* wsb = (char*)d_ws;
    size_t off = 0;
    auto alloc = [&](size_t bytes) -> void* {
        off = (off + 255) & ~(size_t)255;
        void* p = wsb + off;
        off += bytes;
        return p;
    };
    float*    h1     = (float*)   alloc((size_t)N * HC1 * 4);
    float*    o1     = (float*)   alloc((size_t)N * HC1 * 4);  // out1 accum -> x2 in place
    float*    h2     = (float*)   alloc((size_t)N * C2 * 4);
    float*    as1    = (float*)   alloc((size_t)N * H1 * 4);
    float*    ad1    = (float*)   alloc((size_t)N * H1 * 4);
    unsigned* em1    = (unsigned*)alloc((size_t)N * H1 * 4);
    float*    dn1    = (float*)   alloc((size_t)N * H1 * 4);
    float*    as2    = (float*)   alloc((size_t)N * 4);
    float*    ad2    = (float*)   alloc((size_t)N * 4);
    unsigned* em2    = (unsigned*)alloc((size_t)N * 4);
    float*    dn2    = (float*)   alloc((size_t)N * 4);
    unsigned short* w1hi = (unsigned short*)alloc((size_t)m  * HC1 * 2);
    unsigned short* w1lo = (unsigned short*)alloc((size_t)m  * HC1 * 2);
    unsigned short* w2hi = (unsigned short*)alloc((size_t)HC1 * C2 * 2);
    unsigned short* w2lo = (unsigned short*)alloc((size_t)HC1 * C2 * 2);

    const int T = 256;
    auto grid1 = [](size_t n, int t) { return (unsigned)((n + t - 1) / t); };

    // ---- pack weights into WMMA-lane layout ----
    pack_w<<<grid1((size_t)m * HC1, T), T, 0, stream>>>(W1, w1hi, w1lo, m, HC1);
    pack_w<<<grid1((size_t)HC1 * C2, T), T, 0, stream>>>(W2, w2hi, w2lo, HC1, C2);

    // ================= layer 1 =================
    {
        int nunits = (N / 16) * (HC1 / 64);
        gemm_bf16x3<<<grid1(nunits, 8), T, 0, stream>>>(x, w1hi, w1lo, h1, m, HC1, nunits);

        attn_dots<<<grid1(N, 8), T, 0, stream>>>(h1, att_s1, att_d1, as1, ad1, N, H1, Ch1);

        fill_u32<<<grid1((size_t)N * H1, T), T, 0, stream>>>(em1, 0u, (size_t)N * H1);
        fill_u32<<<grid1((size_t)N * H1, T), T, 0, stream>>>((unsigned*)dn1, 0u, (size_t)N * H1);
        fill_u32<<<grid1((size_t)N * HC1, T), T, 0, stream>>>((unsigned*)o1, 0u, (size_t)N * HC1);

        edge_max  <<<grid1(E + N, T), T, 0, stream>>>(ei, as1, ad1, em1, E, N, H1, slope);
        edge_denom<<<grid1(E + N, T), T, 0, stream>>>(ei, as1, ad1, em1, dn1, E, N, H1, slope);
        edge_msg  <<<grid1((size_t)(E + N) * Ch1, T), T, 0, stream>>>(ei, as1, ad1, em1, dn1,
                                                                      h1, o1, E, N, H1, Ch1, slope);
        bias_act<<<grid1((size_t)N * HC1, T), T, 0, stream>>>(o1, bias1, (size_t)N * HC1, HC1, 1);
    }

    // ================= layer 2 =================
    {
        int nunits = (N / 16) * (C2 / 64);
        gemm_bf16x3<<<grid1(nunits, 8), T, 0, stream>>>(o1, w2hi, w2lo, h2, HC1, C2, nunits);

        attn_dots<<<grid1(N, 8), T, 0, stream>>>(h2, att_s2, att_d2, as2, ad2, N, H2, C2);

        fill_u32<<<grid1((size_t)N, T), T, 0, stream>>>(em2, 0u, (size_t)N);
        fill_u32<<<grid1((size_t)N, T), T, 0, stream>>>((unsigned*)dn2, 0u, (size_t)N);
        fill_u32<<<grid1((size_t)N * C2, T), T, 0, stream>>>((unsigned*)d_out, 0u, (size_t)N * C2);

        edge_max  <<<grid1(E + N, T), T, 0, stream>>>(ei, as2, ad2, em2, E, N, H2, slope);
        edge_denom<<<grid1(E + N, T), T, 0, stream>>>(ei, as2, ad2, em2, dn2, E, N, H2, slope);
        edge_msg  <<<grid1((size_t)(E + N) * C2, T), T, 0, stream>>>(ei, as2, ad2, em2, dn2,
                                                                     h2, (float*)d_out, E, N, H2, C2, slope);
        bias_act<<<grid1((size_t)N * C2, T), T, 0, stream>>>((float*)d_out, bias2, (size_t)N * C2, C2, 0);
    }
}
